// EncoderBlock_30270929502811
// MI455X (gfx1250) — matrix operand order repaired and checked by reading, not compiled, not run on hardware
//
#include <hip/hip_runtime.h>

// ---------------------------------------------------------------------------
// Types
// ---------------------------------------------------------------------------
typedef __bf16  v4bf  __attribute__((ext_vector_type(4)));
typedef __bf16  v8bf  __attribute__((ext_vector_type(8)));
typedef __bf16  v16bf __attribute__((ext_vector_type(16)));
typedef float   v8f   __attribute__((ext_vector_type(8)));
typedef int     i32x4_ __attribute__((vector_size(16)));

#define BB 4
#define NN 2048
#define CC 1024
#define HH 8
#define DD 128
#define MM (BB * NN)          // 8192 rows
#define SCALE 0.08838834764831845f
#define EPSV 1e-5f

// CDNA5 async global->LDS copy path (ASYNCcnt), with safe fallback.
#if defined(__has_builtin)
#if __has_builtin(__builtin_amdgcn_global_load_async_to_lds_b128)
#define ASYNC_CP 1
#endif
#endif

__device__ __forceinline__ void cp16(const __bf16* g, __bf16* l) {
#ifdef ASYNC_CP
  __builtin_amdgcn_global_load_async_to_lds_b128(
      (__attribute__((address_space(1))) i32x4_*)(g),
      (__attribute__((address_space(3))) i32x4_*)(l), 0, 0);
#else
  *(v8bf*)l = *(const v8bf*)g;
#endif
}

__device__ __forceinline__ void async_join() {
#ifdef ASYNC_CP
  asm volatile("s_wait_asynccnt 0x0" ::: "memory");
#endif
}

__device__ __forceinline__ v8f vzero8() {
  v8f z = {0.f, 0.f, 0.f, 0.f, 0.f, 0.f, 0.f, 0.f};
  return z;
}

__device__ __forceinline__ v16bf cat8(v8bf a, v8bf b) {
  return __builtin_shufflevector(a, b, 0,1,2,3,4,5,6,7,8,9,10,11,12,13,14,15);
}

// A-fragment (16-bit A 16x32, ISA 7.12.2): two 8-element runs at +0 / +16.
__device__ __forceinline__ v16bf frag_a_runs(const __bf16* p) {
  v8bf lo = *(const v8bf*)p;
  v8bf hi = *(const v8bf*)(p + 16);
  return cat8(lo, hi);
}

// B-fragment: 16 contiguous K elements for this lane's column.
__device__ __forceinline__ v16bf frag_16(const __bf16* p) {
  v8bf lo = *(const v8bf*)p;
  v8bf hi = *(const v8bf*)(p + 8);
  return cat8(lo, hi);
}

__device__ __forceinline__ float eluf(float x) {
  return x > 0.f ? x : (__expf(x) - 1.f);
}

// ---------------------------------------------------------------------------
// fp32 -> bf16 convert (weights)
// ---------------------------------------------------------------------------
__global__ void cvt_bf16_kernel(const float* __restrict__ in,
                                __bf16* __restrict__ out, int n4) {
  int i = blockIdx.x * blockDim.x + threadIdx.x;
  if (i < n4) {
    float4 v = ((const float4*)in)[i];
    v4bf o = {(__bf16)v.x, (__bf16)v.y, (__bf16)v.z, (__bf16)v.w};
    ((v4bf*)out)[i] = o;
  }
}

// ---------------------------------------------------------------------------
// LayerNorm: one 256-thread block per row of C=1024, bf16 output
// ---------------------------------------------------------------------------
__global__ __launch_bounds__(256) void ln_kernel(const float* __restrict__ x,
                                                 const float* __restrict__ g,
                                                 const float* __restrict__ bb,
                                                 __bf16* __restrict__ out) {
  const int row = blockIdx.x, t = threadIdx.x;
  const int lane = t & 31, w = t >> 5;
  const float4 v = ((const float4*)(x + (size_t)row * CC))[t];
  float s  = v.x + v.y + v.z + v.w;
  float s2 = v.x*v.x + v.y*v.y + v.z*v.z + v.w*v.w;
#pragma unroll
  for (int m = 1; m < 32; m <<= 1) {
    s  += __shfl_xor(s,  m, 32);
    s2 += __shfl_xor(s2, m, 32);
  }
  __shared__ float r1[8], r2[8];
  if (lane == 0) { r1[w] = s; r2[w] = s2; }
  __syncthreads();
  float ts = 0.f, ts2 = 0.f;
#pragma unroll
  for (int i = 0; i < 8; i++) { ts += r1[i]; ts2 += r2[i]; }
  const float mu  = ts * (1.f / CC);
  const float var = ts2 * (1.f / CC) - mu * mu;
  const float rs  = rsqrtf(var + EPSV);
  const int c = t * 4;
  v4bf o = {(__bf16)((v.x - mu) * rs * g[c+0] + bb[c+0]),
            (__bf16)((v.y - mu) * rs * g[c+1] + bb[c+1]),
            (__bf16)((v.z - mu) * rs * g[c+2] + bb[c+2]),
            (__bf16)((v.w - mu) * rs * g[c+3] + bb[c+3])};
  *(v4bf*)(out + (size_t)row * CC + c) = o;
}

// ---------------------------------------------------------------------------
// Tiled bf16 WMMA GEMM: C[M,N] = epilogue(A[M,K] @ W[N,K]^T)
// Block tile 128x128, 8 waves (2 M x 4 N), wave tile 64x32 (4x2 WMMA tiles).
// Double-buffered LDS: next tile's async global->LDS DMA overlaps WMMAs.
// MODE 0: qkv split; q scaled -> [B,H,N,D]; k -> [B,H,N,D]; v -> [B,H,D,N]^T
// MODE 1: + bias + residual -> fp32
// MODE 2: ELU(+bias) -> bf16
// MODE 3: ELU(+bias) -> BatchNorm -> + residual -> fp32
// ---------------------------------------------------------------------------
template <int MODE>
__global__ __launch_bounds__(256) void gemm_bf16_kernel(
    const __bf16* __restrict__ A, const __bf16* __restrict__ W,
    int Nn, int K,
    const float* __restrict__ bias, const float* __restrict__ resid,
    float* __restrict__ outf, __bf16* __restrict__ outb,
    __bf16* __restrict__ qo, __bf16* __restrict__ ko, __bf16* __restrict__ vo,
    const float* __restrict__ bng, const float* __restrict__ bnb,
    const float* __restrict__ bnm, const float* __restrict__ bnv) {
  __shared__ __bf16 As[2][128][40];  // +8 pad: 80B row = 20 banks, no conflicts
  __shared__ __bf16 Ws[2][128][40];

  const int tid  = threadIdx.x;
  const int lane = tid & 31, wave = tid >> 5;
  const int wm = wave >> 2, wn = wave & 3;
  const int hi = lane >> 4, ln16 = lane & 15;
  const int mbase = blockIdx.y * 128;
  const int nbase = blockIdx.x * 128;

  v8f acc[4][2];
#pragma unroll
  for (int i = 0; i < 4; i++)
#pragma unroll
    for (int j = 0; j < 2; j++) acc[i][j] = vzero8();

  const int ldr = tid >> 1;           // 0..127
  const int ldc = (tid & 1) * 16;     // 0 or 16

  auto stage = [&](int buf, int kb) {
    const __bf16* sa = A + (size_t)(mbase + ldr) * K + kb + ldc;
    cp16(sa,     &As[buf][ldr][ldc]);
    cp16(sa + 8, &As[buf][ldr][ldc + 8]);
    const __bf16* sw = W + (size_t)(nbase + ldr) * K + kb + ldc;
    cp16(sw,     &Ws[buf][ldr][ldc]);
    cp16(sw + 8, &Ws[buf][ldr][ldc + 8]);
  };

  const int KT = K / 32;
  stage(0, 0);
  for (int ki = 0; ki < KT; ki++) {
    const int cur = ki & 1;
    async_join();        // my cur-tile copies landed in LDS
    __syncthreads();     // everyone's cur tile visible
    if (ki + 1 < KT) stage(1 - cur, (ki + 1) * 32);  // DMA overlaps WMMAs

    v16bf af[4], bf[2];
#pragma unroll
    for (int mi = 0; mi < 4; mi++)
      af[mi] = frag_16(&As[cur][wm * 64 + mi * 16 + ln16][hi * 16]);
#pragma unroll
    for (int ni = 0; ni < 2; ni++)
      bf[ni] = frag_16(&Ws[cur][wn * 32 + ni * 16 + ln16][hi * 16]);

#pragma unroll
    for (int mi = 0; mi < 4; mi++)
#pragma unroll
      for (int ni = 0; ni < 2; ni++)
        acc[mi][ni] = __builtin_amdgcn_wmma_f32_16x16x32_bf16(
            false, af[mi], false, bf[ni], (short)0, acc[mi][ni], false, false);
    __syncthreads();     // all reads of cur done before it is restaged
  }

  // ---- epilogue ----
#pragma unroll
  for (int mi = 0; mi < 4; mi++) {
#pragma unroll
    for (int ni = 0; ni < 2; ni++) {
      const int rowb = mbase + wm * 64 + mi * 16 + 8 * hi;  // 8 consecutive rows
      const int col  = nbase + wn * 32 + ni * 16 + ln16;
      if (MODE == 0) {
        const int which = col >> 10, hh = (col >> 7) & 7, d = col & 127;
        const int b2 = rowb >> 11, n0 = rowb & (NN - 1);
        if (which == 2) {
          // V stored transposed [B,H,D,N]: rows are consecutive n -> v8bf store
          v8bf pk;
#pragma unroll
          for (int r = 0; r < 8; r++) pk[r] = (__bf16)acc[mi][ni][r];
          *(v8bf*)(vo + ((((size_t)b2 * HH + hh) * DD + d) * NN + n0)) = pk;
        } else {
          __bf16* dst = (which == 0) ? qo : ko;
          const float sc = (which == 0) ? SCALE : 1.f;
#pragma unroll
          for (int r = 0; r < 8; r++)
            dst[(((size_t)b2 * HH + hh) * NN + n0 + r) * DD + d] =
                (__bf16)(acc[mi][ni][r] * sc);
        }
      } else if (MODE == 1) {
#pragma unroll
        for (int r = 0; r < 8; r++) {
          const size_t idx = (size_t)(rowb + r) * CC + col;
          outf[idx] = acc[mi][ni][r] + bias[col] + resid[idx];
        }
      } else if (MODE == 2) {
#pragma unroll
        for (int r = 0; r < 8; r++)
          outb[(size_t)(rowb + r) * CC + col] =
              (__bf16)eluf(acc[mi][ni][r] + bias[col]);
      } else {  // MODE 3
#pragma unroll
        for (int r = 0; r < 8; r++) {
          float t = eluf(acc[mi][ni][r] + bias[col]);
          t = (t - bnm[col]) * rsqrtf(bnv[col] + EPSV) * bng[col] + bnb[col];
          const size_t idx = (size_t)(rowb + r) * CC + col;
          outf[idx] = t + resid[idx];
        }
      }
    }
  }
}

// ---------------------------------------------------------------------------
// Flash attention: one WG per (b, h, 128-row q tile); 8 waves x 16 q-rows.
// 64-key blocks, double-buffered K/V^T staging (async DMA overlaps compute):
// 16 WMMAs for S = Q@K^T (scale baked into Q), online softmax with max via
// shfl butterfly and ROW-SUM VIA WMMA against an all-ones B matrix
// (replicated-lane layout matches lrow), P restaged through per-wave LDS,
// 16 WMMAs for O += P@V using pre-transposed V [B,H,D,N].
// ---------------------------------------------------------------------------
__global__ __launch_bounds__(256) void attn_kernel(
    const __bf16* __restrict__ q, const __bf16* __restrict__ k,
    const __bf16* __restrict__ v, __bf16* __restrict__ o) {
  __shared__ __bf16 Ks[2][64][136];    // [key][d]   272B row -> conflict-free
  __shared__ __bf16 Vts[2][128][72];   // V^T: [d][key] 144B row -> conflict-free
  __shared__ __bf16 Ps[8][16][72];     // per-wave P staging

  const int tid = threadIdx.x, lane = tid & 31, w = tid >> 5;
  const int hi = lane >> 4, ln16 = lane & 15;
  const int qt = blockIdx.x, h = blockIdx.y, b = blockIdx.z;
  const int qbase = qt * 128;
  const size_t headoff = ((size_t)b * HH + h) * (size_t)NN * DD;
  const __bf16* qh  = q + headoff;
  const __bf16* kh  = k + headoff;
  const __bf16* vth = v + headoff;  // [D][N] layout

  // preload Q fragments for this wave's 16 rows (D=128 -> 4 k-steps)
  v16bf qf[4];
  {
    const __bf16* qp = qh + (size_t)(qbase + w * 16 + ln16) * DD;
#pragma unroll
    for (int kt = 0; kt < 4; kt++) qf[kt] = frag_16(qp + kt * 32 + hi * 16);
  }
  v16bf onesf;
  {
    v8bf o1;
#pragma unroll
    for (int i = 0; i < 8; i++) o1[i] = (__bf16)1.f;
    onesf = cat8(o1, o1);
  }

  float mrow[8], lrow[8], alpha[8];
  v8f oacc[8];
#pragma unroll
  for (int r = 0; r < 8; r++) { mrow[r] = -3e38f; lrow[r] = 0.f; }
#pragma unroll
  for (int nt = 0; nt < 8; nt++) oacc[nt] = vzero8();

  const int kk = tid >> 2;         // 0..63 : key row for K staging
  const int kd = (tid & 3) * 32;   // d segment
  const int vd = tid >> 1;         // 0..127: d row for V^T staging
  const int vk = (tid & 1) * 32;   // key segment

  auto stageKV = [&](int buf, int jb) {
    const __bf16* ks = kh + (size_t)(jb * 64 + kk) * DD + kd;
#pragma unroll
    for (int i = 0; i < 4; i++) cp16(ks + i * 8, &Ks[buf][kk][kd + i * 8]);
    const __bf16* vs = vth + (size_t)vd * NN + jb * 64 + vk;
#pragma unroll
    for (int i = 0; i < 4; i++) cp16(vs + i * 8, &Vts[buf][vd][vk + i * 8]);
  };

  const int JT = NN / 64;
  stageKV(0, 0);
  for (int jb = 0; jb < JT; jb++) {
    const int cur = jb & 1;
    async_join();        // my cur K/V tile copies landed
    __syncthreads();     // everyone's cur tile visible
    if (jb + 1 < JT) stageKV(1 - cur, jb + 1);  // DMA overlaps compute

    // S = Q @ K^T over D=128 (4 x k32), four 16-key tiles
    v8f s[4];
#pragma unroll
    for (int j = 0; j < 4; j++) s[j] = vzero8();
#pragma unroll
    for (int kt = 0; kt < 4; kt++) {
#pragma unroll
      for (int j = 0; j < 4; j++) {
        v16bf kf = frag_16(&Ks[cur][j * 16 + ln16][kt * 32 + hi * 16]);
        s[j] = __builtin_amdgcn_wmma_f32_16x16x32_bf16(
            false, qf[kt], false, kf, (short)0, s[j], false, false);
      }
    }

    // online softmax: running max via butterfly, exp -> bf16 P in LDS
#pragma unroll
    for (int r = 0; r < 8; r++) {
      float mx = fmaxf(fmaxf(s[0][r], s[1][r]), fmaxf(s[2][r], s[3][r]));
      mx = fmaxf(mx, __shfl_xor(mx, 1, 16));
      mx = fmaxf(mx, __shfl_xor(mx, 2, 16));
      mx = fmaxf(mx, __shfl_xor(mx, 4, 16));
      mx = fmaxf(mx, __shfl_xor(mx, 8, 16));
      const float mnew = fmaxf(mrow[r], mx);
      alpha[r] = __expf(mrow[r] - mnew);
      mrow[r]  = mnew;
#pragma unroll
      for (int j = 0; j < 4; j++)
        Ps[w][r + 8 * hi][j * 16 + ln16] = (__bf16)__expf(s[j][r] - mnew);
    }
#pragma unroll
    for (int nt = 0; nt < 8; nt++)
#pragma unroll
      for (int r = 0; r < 8; r++) oacc[nt][r] *= alpha[r];

    asm volatile("s_wait_dscnt 0" ::: "memory");
    v16bf pf0 = frag_16(&Ps[w][ln16][hi * 16]);
    v16bf pf1 = frag_16(&Ps[w][ln16][32 + hi * 16]);

    // row-sum of P via WMMA with all-ones B: every lane gets its row's sum
    v8f lb = __builtin_amdgcn_wmma_f32_16x16x32_bf16(
        false, pf0, false, onesf, (short)0, vzero8(), false, false);
    lb = __builtin_amdgcn_wmma_f32_16x16x32_bf16(
        false, pf1, false, onesf, (short)0, lb, false, false);
#pragma unroll
    for (int r = 0; r < 8; r++) lrow[r] = lrow[r] * alpha[r] + lb[r];

    // O += P @ V  (contraction over 64 keys = 2 x k32)
#pragma unroll
    for (int nt = 0; nt < 8; nt++) {
      v16bf vf0 = frag_16(&Vts[cur][nt * 16 + ln16][hi * 16]);
      v16bf vf1 = frag_16(&Vts[cur][nt * 16 + ln16][32 + hi * 16]);
      oacc[nt] = __builtin_amdgcn_wmma_f32_16x16x32_bf16(
          false, pf0, false, vf0, (short)0, oacc[nt], false, false);
      oacc[nt] = __builtin_amdgcn_wmma_f32_16x16x32_bf16(
          false, pf1, false, vf1, (short)0, oacc[nt], false, false);
    }
    __syncthreads();     // all reads of cur done before it is restaged
  }

  // O /= l ; write [B,N,H*D] bf16
#pragma unroll
  for (int nt = 0; nt < 8; nt++) {
#pragma unroll
    for (int r = 0; r < 8; r++) {
      const float val = oacc[nt][r] / lrow[r];
      const int row = qbase + w * 16 + r + 8 * hi;
      o[((size_t)(b * NN + row)) * CC + h * DD + nt * 16 + ln16] = (__bf16)val;
    }
  }
}

// ---------------------------------------------------------------------------
// Launcher
// ---------------------------------------------------------------------------
extern "C" void kernel_launch(void* const* d_in, const int* in_sizes, int n_in,
                              void* d_out, int out_size, void* d_ws, size_t ws_size,
                              hipStream_t stream) {
  const float* x      = (const float*)d_in[0];
  const float* ln1_g  = (const float*)d_in[1];
  const float* ln1_b  = (const float*)d_in[2];
  const float* w_qkv  = (const float*)d_in[3];
  const float* w_proj = (const float*)d_in[4];
  const float* b_proj = (const float*)d_in[5];
  const float* ln2_g  = (const float*)d_in[6];
  const float* ln2_b  = (const float*)d_in[7];
  const float* w1     = (const float*)d_in[8];
  const float* b1     = (const float*)d_in[9];
  const float* w2     = (const float*)d_in[10];
  const float* b2     = (const float*)d_in[11];
  const float* bn_g   = (const float*)d_in[12];
  const float* bn_b   = (const float*)d_in[13];
  const float* bn_m   = (const float*)d_in[14];
  const float* bn_v   = (const float*)d_in[15];
  float* out = (float*)d_out;

  // workspace carve-up (256B aligned)
  char* ws = (char*)d_ws;
  size_t off = 0;
  auto carve = [&](size_t bytes) {
    char* p = ws + off;
    off = (off + bytes + 255) & ~(size_t)255;
    return p;
  };
  __bf16* wqkv_bf  = (__bf16*)carve((size_t)3 * CC * CC * 2);
  __bf16* wproj_bf = (__bf16*)carve((size_t)CC * CC * 2);
  __bf16* w1_bf    = (__bf16*)carve((size_t)CC * CC * 2);
  __bf16* w2_bf    = (__bf16*)carve((size_t)CC * CC * 2);
  __bf16* hA  = (__bf16*)carve((size_t)MM * CC * 2);  // LN1 out, later LN2 out
  __bf16* qb  = (__bf16*)carve((size_t)MM * CC * 2);  // [B,H,N,D], scaled
  __bf16* kb  = (__bf16*)carve((size_t)MM * CC * 2);  // [B,H,N,D]
  __bf16* vb  = (__bf16*)carve((size_t)MM * CC * 2);  // [B,H,D,N] (transposed)
  __bf16* ob  = (__bf16*)carve((size_t)MM * CC * 2);  // attn out, later MLP mid
  float*  x1  = (float*)carve((size_t)MM * CC * 4);   // x + attn branch
  (void)ws_size; (void)n_in; (void)in_sizes; (void)out_size;

  // 1) weights -> bf16
  {
    int n4;
    n4 = 3 * CC * CC / 4;
    cvt_bf16_kernel<<<(n4 + 255) / 256, 256, 0, stream>>>(w_qkv, wqkv_bf, n4);
    n4 = CC * CC / 4;
    cvt_bf16_kernel<<<(n4 + 255) / 256, 256, 0, stream>>>(w_proj, wproj_bf, n4);
    cvt_bf16_kernel<<<(n4 + 255) / 256, 256, 0, stream>>>(w1, w1_bf, n4);
    cvt_bf16_kernel<<<(n4 + 255) / 256, 256, 0, stream>>>(w2, w2_bf, n4);
  }

  // 2) LN1
  ln_kernel<<<MM, 256, 0, stream>>>(x, ln1_g, ln1_b, hA);

  // 3) QKV projection -> q (scaled), k [B,H,N,D]; v transposed [B,H,D,N]
  gemm_bf16_kernel<0><<<dim3(3 * CC / 128, MM / 128), 256, 0, stream>>>(
      hA, wqkv_bf, 3 * CC, CC, nullptr, nullptr, nullptr, nullptr,
      qb, kb, vb, nullptr, nullptr, nullptr, nullptr);

  // 4) flash attention -> ob bf16 [B,N,C]
  attn_kernel<<<dim3(NN / 128, HH, BB), 256, 0, stream>>>(qb, kb, vb, ob);

  // 5) output projection + bias + residual -> x1 fp32
  gemm_bf16_kernel<1><<<dim3(CC / 128, MM / 128), 256, 0, stream>>>(
      ob, wproj_bf, CC, CC, b_proj, x, x1, nullptr,
      nullptr, nullptr, nullptr, nullptr, nullptr, nullptr, nullptr);

  // 6) LN2
  ln_kernel<<<MM, 256, 0, stream>>>(x1, ln2_g, ln2_b, hA);

  // 7) MLP fc1 + ELU -> ob bf16
  gemm_bf16_kernel<2><<<dim3(CC / 128, MM / 128), 256, 0, stream>>>(
      hA, w1_bf, CC, CC, b1, nullptr, nullptr, ob,
      nullptr, nullptr, nullptr, nullptr, nullptr, nullptr, nullptr);

  // 8) MLP fc2 + ELU + BN + residual -> out fp32
  gemm_bf16_kernel<3><<<dim3(CC / 128, MM / 128), 256, 0, stream>>>(
      ob, w2_bf, CC, CC, b2, x1, out, nullptr,
      nullptr, nullptr, nullptr, bn_g, bn_b, bn_m, bn_v);
}
